// LandauerLanguageModel_41437844471864
// MI455X (gfx1250) — matrix operand order repaired and checked by reading, not compiled, not run on hardware
//
#include <hip/hip_runtime.h>
#include <hip/hip_bf16.h>
#include <stdint.h>

#define DEV __device__ __forceinline__

constexpr int Bb = 64;    // batch
constexpr int Ss = 512;   // sequence length
constexpr int Ee = 512;   // embedding dim
constexpr int Hh = 1024;  // hidden dim
constexpr int Vv = 256;   // vocab
constexpr int Ll = 3;     // layers

typedef __attribute__((ext_vector_type(16))) __bf16 bf16x16;
typedef __attribute__((ext_vector_type(8)))  float  f32x8;

union Frag {
  uint32_t u[8];
  uint4    q[2];
  bf16x16  v;
};

// bf16 with round-to-nearest-even (used off the hot path only)
DEV uint32_t f2bf_rne(float f) {
  union { float f; uint32_t u; } x; x.f = f;
  uint32_t r = x.u + 0x7FFFu + ((x.u >> 16) & 1u);
  return r >> 16;
}
DEV uint32_t pack2_rne(float lo, float hi) {
  return (f2bf_rne(lo) & 0xFFFFu) | (f2bf_rne(hi) << 16);
}
// bf16 truncation pack: one v_perm_b32 (hot path: A-matrix conversion)
DEV uint32_t pack2_t(float lo, float hi) {
  return __builtin_amdgcn_perm(__float_as_uint(hi), __float_as_uint(lo),
                               0x07060302u);
}
DEV float fast_tanh(float x) {
#if __has_builtin(__builtin_amdgcn_tanhf)
  return __builtin_amdgcn_tanhf(x);   // v_tanh_f32 (new in gfx1250)
#else
  return tanhf(x);
#endif
}

// ---------------------------------------------------------------------------
// Pack an f32 weight matrix W (K x N, row major) into bf16 WMMA B-fragments.
// Tile = (32 K) x (16 N). Linear tile id = nt*(K/32)+kc (k-loop streams
// contiguously). Within a tile: [lane][8 dwords].
// B-layout (v_wmma_f32_16x16x32_bf16): N = lane&15;
//   lanes 0-15 : K = 0..7 (d0..3 pairs) and 16..23 (d4..7)
//   lanes 16-31: K = 8..15 (d0..3 pairs) and 24..31 (d4..7)
// ---------------------------------------------------------------------------
__global__ __launch_bounds__(256) void pack_w_kernel(
    const float* __restrict__ W, uint32_t* __restrict__ Wp, int K, int N)
{
  const int lane = threadIdx.x & 31;
  const int wave = threadIdx.x >> 5;
  const int kchunks = K >> 5;
  const int ntiles  = (N >> 4) * kchunks;
  const int tile = blockIdx.x * 8 + wave;
  if (tile >= ntiles) return;
  const int nt = tile / kchunks;
  const int kc = tile - nt * kchunks;
  const int n  = nt * 16 + (lane & 15);
  const int kb = kc * 32 + ((lane >= 16) ? 8 : 0);
  uint32_t out[8];
#pragma unroll
  for (int d = 0; d < 8; ++d) {
    const int k = kb + 2 * d + ((d >= 4) ? 8 : 0);
    out[d] = pack2_rne(W[(size_t)k * N + n], W[(size_t)(k + 1) * N + n]);
  }
  uint32_t* dst = Wp + ((size_t)tile * 32 + lane) * 8;
#pragma unroll
  for (int d = 0; d < 8; ++d) dst[d] = out[d];
}

// ---------------------------------------------------------------------------
// Fragment loaders.
// A-layout (16-bit A 16x32): M = lane&15;
//   lanes 0-15 : K = +0..7 (v0..3 pairs) and +16..23 (v4..7)
//   lanes 16-31: K = +8..15            and +24..31
// ---------------------------------------------------------------------------
DEV void load_b_frag(Frag& b, const uint32_t* __restrict__ wtile, int kc) {
  const uint32_t* wp = wtile + (size_t)kc * 256u;
  b.q[0] = *(const uint4*)(wp);
  b.q[1] = *(const uint4*)(wp + 4);
}
DEV void load_a_f32(Frag& a, const float* __restrict__ arow, int k0, int kSel) {
  const float* ap = arow + k0 + kSel;
  const float4 f0 = *(const float4*)(ap);
  const float4 f1 = *(const float4*)(ap + 4);
  const float4 f2 = *(const float4*)(ap + 16);
  const float4 f3 = *(const float4*)(ap + 20);
  a.u[0] = pack2_t(f0.x, f0.y); a.u[1] = pack2_t(f0.z, f0.w);
  a.u[2] = pack2_t(f1.x, f1.y); a.u[3] = pack2_t(f1.z, f1.w);
  a.u[4] = pack2_t(f2.x, f2.y); a.u[5] = pack2_t(f2.z, f2.w);
  a.u[6] = pack2_t(f3.x, f3.y); a.u[7] = pack2_t(f3.z, f3.w);
}
DEV void load_a_bf16(Frag& a, const unsigned short* __restrict__ arow,
                     int k0, int kSel) {
  const unsigned short* ap = arow + k0 + kSel;
  a.q[0] = *(const uint4*)(ap);        // K+0..7   (pairs)
  a.q[1] = *(const uint4*)(ap + 16);   // K+16..23 (pairs)
}

DEV f32x8 wmma_bf16(const Frag& a, const Frag& b, f32x8 c) {
  return __builtin_amdgcn_wmma_f32_16x16x32_bf16(
      false, a.v, false, b.v, (short)0, c, false, false);
}

// ---------------------------------------------------------------------------
// One RNN layer for one timestep:
//   h_next = tanh(A @ W + b + h_prev * cos(phase*n + h_prev) * scale)
// A is (64 x K):
//   AMODE 0: rows gathered from emb (f32) via idx[:, t]   (layer 0)
//   AMODE 1: bf16 row-major activation buffer             (layers 1,2)
// Each wave owns one 16x16 tile; 8 waves/block -> 64x32 block tile;
// grid.x = H/32. Double-buffered k-loop (kchunks is even).
// ---------------------------------------------------------------------------
template <int AMODE, bool WRITE_ABUF, bool WRITE_TRACE>
__global__ __launch_bounds__(256) void layer_step_kernel(
    const void* __restrict__ Asrc,
    const int*  __restrict__ idx, int t, int K,
    const uint32_t* __restrict__ Wp,
    const float* __restrict__ bias,
    const float* __restrict__ hscale,
    float phase,
    float* __restrict__ state,               // (B,H) f32: read h_prev / write h_next
    unsigned short* __restrict__ abuf_out,   // (B,H) bf16 (next layer's A)
    float* __restrict__ trace)               // (B,S,H) f32
{
  const int lane = threadIdx.x & 31;
  const int wave = threadIdx.x >> 5;
  const int m0 = (wave & 3) * 16;                 // 4 M-tiles cover B=64
  const int n0 = blockIdx.x * 32 + (wave >> 2) * 16;

  const int mrow = m0 + (lane & 15);              // A row (batch index)
  const int kSel = (lane >= 16) ? 8 : 0;

  const float*          arow_f = nullptr;
  const unsigned short* arow_b = nullptr;
  if constexpr (AMODE == 0) {
    const int r = idx[(size_t)mrow * Ss + t];
    arow_f = (const float*)Asrc + (size_t)r * Ee;
  } else {
    arow_b = (const unsigned short*)Asrc + (size_t)mrow * K;
  }

  const int kchunks = K >> 5;
  const uint32_t* wtile =
      Wp + ((size_t)(n0 >> 4) * kchunks) * 256u + (size_t)lane * 8u;

  Frag a0, b0, a1, b1;
  f32x8 acc = {0.f, 0.f, 0.f, 0.f, 0.f, 0.f, 0.f, 0.f};

  load_b_frag(b0, wtile, 0);
  if constexpr (AMODE == 0) load_a_f32(a0, arow_f, 0, kSel);
  else                      load_a_bf16(a0, arow_b, 0, kSel);

  for (int kc = 0; kc < kchunks; kc += 2) {
    load_b_frag(b1, wtile, kc + 1);
    if constexpr (AMODE == 0) load_a_f32(a1, arow_f, (kc + 1) * 32, kSel);
    else                      load_a_bf16(a1, arow_b, (kc + 1) * 32, kSel);
    __builtin_prefetch((const void*)(wtile + (size_t)(kc + 2) * 256u), 0, 0);
    acc = wmma_bf16(a0, b0, acc);

    if (kc + 2 < kchunks) {
      load_b_frag(b0, wtile, kc + 2);
      if constexpr (AMODE == 0) load_a_f32(a0, arow_f, (kc + 2) * 32, kSel);
      else                      load_a_bf16(a0, arow_b, (kc + 2) * 32, kSel);
    }
    acc = wmma_bf16(a1, b1, acc);
  }

  // Epilogue: C/D layout -> n = lane&15, m = vgpr + 8*(lane>=16)
  const int n  = n0 + (lane & 15);
  const int mB = m0 + ((lane >= 16) ? 8 : 0);
  const float bv = bias[n];
  const float sv = hscale[n];
  const float pn = phase * (float)n;
#pragma unroll
  for (int r = 0; r < 8; ++r) {
    const int m = mB + r;
    const size_t off = (size_t)m * Hh + n;
    const float hp  = state[off];
    const float rot = hp * __cosf(pn + hp) * sv;
    const float val = fast_tanh(acc[r] + bv + rot);
    state[off] = val;
    if constexpr (WRITE_ABUF) abuf_out[off] = (unsigned short)f2bf_rne(val);
    if constexpr (WRITE_TRACE) trace[((size_t)m * Ss + t) * Hh + n] = val;
  }
}

// ---------------------------------------------------------------------------
// logits = trace(32768 x 1024, f32) @ wout(1024 x 256, packed bf16) + bout
// Block = 8 waves stacked in M (128 x 16 tile). Grid = (V/16, 32768/128).
// ---------------------------------------------------------------------------
__global__ __launch_bounds__(256) void logits_kernel(
    const float* __restrict__ trace,
    const uint32_t* __restrict__ Wp,
    const float* __restrict__ bout,
    float* __restrict__ logits)
{
  const int lane = threadIdx.x & 31;
  const int wave = threadIdx.x >> 5;
  const int m0 = (blockIdx.y * 8 + wave) * 16;
  const int n0 = blockIdx.x * 16;
  const int mrow = m0 + (lane & 15);
  const int kSel = (lane >= 16) ? 8 : 0;

  const float* arow = trace + (size_t)mrow * Hh;
  const int kchunks = Hh >> 5;  // 32 (even)
  const uint32_t* wtile =
      Wp + ((size_t)(n0 >> 4) * kchunks) * 256u + (size_t)lane * 8u;

  Frag a0, b0, a1, b1;
  f32x8 acc = {0.f, 0.f, 0.f, 0.f, 0.f, 0.f, 0.f, 0.f};

  load_b_frag(b0, wtile, 0);
  load_a_f32(a0, arow, 0, kSel);

  for (int kc = 0; kc < kchunks; kc += 2) {
    load_b_frag(b1, wtile, kc + 1);
    load_a_f32(a1, arow, (kc + 1) * 32, kSel);
    acc = wmma_bf16(a0, b0, acc);

    if (kc + 2 < kchunks) {
      load_b_frag(b0, wtile, kc + 2);
      load_a_f32(a0, arow, (kc + 2) * 32, kSel);
    }
    acc = wmma_bf16(a1, b1, acc);
  }

  const int n  = n0 + (lane & 15);
  const int mB = m0 + ((lane >= 16) ? 8 : 0);
  const float bv = bout[n];
#pragma unroll
  for (int r = 0; r < 8; ++r)
    logits[(size_t)(mB + r) * Vv + n] = acc[r] + bv;
}

// ---------------------------------------------------------------------------
extern "C" void kernel_launch(void* const* d_in, const int* in_sizes, int n_in,
                              void* d_out, int out_size, void* d_ws, size_t ws_size,
                              hipStream_t stream) {
  const int*   idx  = (const int*)  d_in[0];   // (B,S)
  const float* emb  = (const float*)d_in[1];   // (V,E)
  const float* w0   = (const float*)d_in[2];   // (E,H)
  const float* b0   = (const float*)d_in[3];
  const float* w1   = (const float*)d_in[4];   // (H,H)
  const float* b1   = (const float*)d_in[5];
  const float* w2   = (const float*)d_in[6];   // (H,H)
  const float* b2   = (const float*)d_in[7];
  const float* hs   = (const float*)d_in[8];   // (L,H)
  const float* wout = (const float*)d_in[9];   // (H,V)
  const float* bout = (const float*)d_in[10];  // (V,)

  float* logits = (float*)d_out;                          // B*S*V
  float* trace  = logits + (size_t)Bb * Ss * Vv;          // B*S*H
  float* hfin   = trace  + (size_t)Bb * Ss * Hh;          // L*B*H (also state)

  char* ws = (char*)d_ws;
  uint32_t* w0p = (uint32_t*)(ws);                         // 1.0 MB
  uint32_t* w1p = (uint32_t*)(ws + (1u << 20));            // 2.0 MB
  uint32_t* w2p = (uint32_t*)(ws + 3u * (1u << 20));       // 2.0 MB
  uint32_t* wop = (uint32_t*)(ws + 5u * (1u << 20));       // 0.5 MB
  unsigned short* ab0 = (unsigned short*)(ws + 5u * (1u << 20) + (512u << 10));
  unsigned short* ab1 = (unsigned short*)(ws + 5u * (1u << 20) + (512u << 10) + (128u << 10));

  // Recurrent state lives in the h_final output region; must start at zero.
  hipMemsetAsync(hfin, 0, (size_t)Ll * Bb * Hh * sizeof(float), stream);

  // Pack weights to bf16 WMMA-B fragment layout (L2-resident, 5 MB total).
  pack_w_kernel<<<128, 256, 0, stream>>>(w0,   w0p, Ee, Hh);  // 1024 tiles
  pack_w_kernel<<<256, 256, 0, stream>>>(w1,   w1p, Hh, Hh);  // 2048 tiles
  pack_w_kernel<<<256, 256, 0, stream>>>(w2,   w2p, Hh, Hh);  // 2048 tiles
  pack_w_kernel<<< 64, 256, 0, stream>>>(wout, wop, Hh, Vv);  //  512 tiles

  const float PHI = 2.3999632297286535f;
  float* s0 = hfin + 0 * (size_t)Bb * Hh;
  float* s1 = hfin + 1 * (size_t)Bb * Hh;
  float* s2 = hfin + 2 * (size_t)Bb * Hh;

  for (int t = 0; t < Ss; ++t) {
    layer_step_kernel<0, true,  false><<<Hh / 32, 256, 0, stream>>>(
        emb, idx, t, Ee, w0p, b0, hs + 0 * Hh, PHI,           s0, ab0, nullptr);
    layer_step_kernel<1, true,  false><<<Hh / 32, 256, 0, stream>>>(
        ab0, nullptr, t, Hh, w1p, b1, hs + 1 * Hh, PHI * 0.5f,  s1, ab1, nullptr);
    layer_step_kernel<1, false, true ><<<Hh / 32, 256, 0, stream>>>(
        ab1, nullptr, t, Hh, w2p, b2, hs + 2 * Hh, PHI * 0.25f, s2, nullptr, trace);
  }

  logits_kernel<<<dim3(Vv / 16, (Bb * Ss) / 128), 256, 0, stream>>>(
      trace, wop, bout, logits);
}
